// Block_14705968021651
// MI455X (gfx1250) — matrix operand order repaired and checked by reading, not compiled, not run on hardware
//
#include <hip/hip_runtime.h>
#include <cstdint>

typedef __bf16 bf16;
typedef __bf16 v16bf __attribute__((ext_vector_type(16)));
typedef float  v8f   __attribute__((ext_vector_type(8)));

union FragBF { v16bf v; uint4 u[2]; bf16 e[16]; };

__device__ __forceinline__ v8f wmma_bf16(v16bf a, v16bf b, v8f c) {
  // D = A(16x32 bf16) * B(32x16 bf16) + C(16x16 f32)
  return __builtin_amdgcn_wmma_f32_16x16x32_bf16(false, a, false, b, (short)0, c,
                                                 false, false);
}

// CDNA5 async global->LDS DMA: copies 16B/lane straight into LDS, tracked by
// ASYNCcnt (no VGPR data path). vdst = LDS byte address, vaddr = 64-bit global.
__device__ __forceinline__ void async_ld_b128(uint32_t lds_addr, const void* gptr) {
  asm volatile("global_load_async_to_lds_b128 %0, %1, off"
               :: "v"(lds_addr), "v"(gptr)
               : "memory");
}
__device__ __forceinline__ void wait_async0() {
  asm volatile("s_wait_asynccnt 0x0" ::: "memory");
}
__device__ __forceinline__ uint32_t lds_addr_of(const void* p) {
  // flat LDS aperture: low 32 bits are the LDS byte address
  return (uint32_t)(uintptr_t)p;
}

// ------------------------------------------------- f32 -> bf16 + transpose
// out[n][k] = (bf16) in[k][n];  32x32 LDS-tiled (K,N multiples of 32)
__global__ __launch_bounds__(256) void transpose_bf16_kernel(
    const float* __restrict__ in, bf16* __restrict__ out, int K, int N) {
  __shared__ float t[32][33];
  const int k0 = blockIdx.y * 32, n0 = blockIdx.x * 32;
  const int tx = threadIdx.x & 31, ty = threadIdx.x >> 5;  // 32 x 8
#pragma unroll
  for (int i = 0; i < 32; i += 8)
    t[ty + i][tx] = in[(size_t)(k0 + ty + i) * N + n0 + tx];
  __syncthreads();
#pragma unroll
  for (int i = 0; i < 32; i += 8)
    out[(size_t)(n0 + ty + i) * K + k0 + tx] = (bf16)t[tx][ty + i];
}

// ---------------------------------------------------------------- layernorm
__global__ __launch_bounds__(256) void layernorm_kernel(
    const float* __restrict__ x, const float* __restrict__ w,
    const float* __restrict__ b, bf16* __restrict__ out, int C) {
  const int row = blockIdx.x;
  const int tid = threadIdx.x;
  const float* xr = x + (size_t)row * C;
  float s = 0.f, s2 = 0.f;
  for (int i = tid; i < C; i += 256) { float v = xr[i]; s += v; s2 += v * v; }
  __shared__ float rs[256], rs2[256];
  rs[tid] = s; rs2[tid] = s2;
  __syncthreads();
  for (int off = 128; off > 0; off >>= 1) {
    if (tid < off) { rs[tid] += rs[tid + off]; rs2[tid] += rs2[tid + off]; }
    __syncthreads();
  }
  const float mu  = rs[0] / C;
  const float var = rs2[0] / C - mu * mu;
  const float inv = rsqrtf(var + 1e-5f);
  bf16* orow = out + (size_t)row * C;
  for (int i = tid; i < C; i += 256)
    orow[i] = (bf16)((xr[i] - mu) * inv * w[i] + b[i]);
}

// ---------------------------------------------------------------- WMMA GEMM
// C[M,N] = A[M,K] @ BT[N,K]^T (+bias / +residual / GELU per EPI)
// 128x128 block tile, 8 wave32s (4M x 2N), wave tile 32x64.
// Double-buffered LDS fed by async global->LDS DMA; one barrier per K-step.
enum { EPI_BIAS_BF16 = 0, EPI_BIAS_RES_F32 = 1, EPI_BIAS_GELU_BF16 = 2 };

template <int EPI>
__global__ __launch_bounds__(256) void gemm_bf16_kernel(
    const bf16* __restrict__ A, const bf16* __restrict__ BT,
    const float* __restrict__ bias, const float* __restrict__ resid,
    void* __restrict__ out, int M, int N, int K) {
  __shared__ __align__(16) bf16 As[2][128][32];  // [m][k]
  __shared__ __align__(16) bf16 Bs[2][128][32];  // [n][k]
  const int tid  = threadIdx.x;
  const int wid  = tid >> 5;
  const int lane = tid & 31;
  const int l    = lane & 15;
  const int hi   = lane >> 4;
  const int m0   = blockIdx.y * 128;
  const int n0   = blockIdx.x * 128;
  const int wm   = (wid & 3) * 32;
  const int wn   = (wid >> 2) * 64;

  v8f acc[2][4] = {};

  // staging pattern: thread t copies 16 bf16 (two async b128) of row t>>1
  const int sr = tid >> 1, sc = (tid & 1) * 16;
  const bf16* aG = A  + (size_t)(m0 + sr) * K + sc;
  const bf16* bG = BT + (size_t)(n0 + sr) * K + sc;

  auto stage = [&](int buf, int k0) {
    const uint32_t la = lds_addr_of(&As[buf][sr][sc]);
    const uint32_t lb = lds_addr_of(&Bs[buf][sr][sc]);
    async_ld_b128(la,      aG + k0);
    async_ld_b128(la + 16, aG + k0 + 8);
    async_ld_b128(lb,      bG + k0);
    async_ld_b128(lb + 16, bG + k0 + 8);
  };

  stage(0, 0);
  const int KT = K / 32;
  for (int kt = 0; kt < KT; ++kt) {
    wait_async0();          // own async copies for buf done
    __syncthreads();        // everyone's copies done; prior reads of other buf done
    if (kt + 1 < KT) stage((kt + 1) & 1, (kt + 1) * 32);
    const int buf = kt & 1;

    FragBF af[2], bfr[4];
#pragma unroll
    for (int mt = 0; mt < 2; ++mt) {
      const bf16* p = &As[buf][wm + mt * 16 + l][0];
      af[mt].u[0] = *reinterpret_cast<const uint4*>(p + hi * 8);
      af[mt].u[1] = *reinterpret_cast<const uint4*>(p + 16 + hi * 8);
    }
#pragma unroll
    for (int nt = 0; nt < 4; ++nt) {
      const bf16* p = &Bs[buf][wn + nt * 16 + l][hi * 8];
      bfr[nt].u[0] = reinterpret_cast<const uint4*>(p)[0];
      bfr[nt].u[1] = reinterpret_cast<const uint4*>(p)[2];
    }
#pragma unroll
    for (int mt = 0; mt < 2; ++mt)
#pragma unroll
      for (int nt = 0; nt < 4; ++nt)
        acc[mt][nt] = wmma_bf16(af[mt].v, bfr[nt].v, acc[mt][nt]);
  }

  // epilogue; C/D layout: VGPR r, lanes 0-15 -> M=r, lanes 16-31 -> M=r+8
#pragma unroll
  for (int mt = 0; mt < 2; ++mt)
#pragma unroll
    for (int nt = 0; nt < 4; ++nt) {
      const int gn = n0 + wn + nt * 16 + l;
      const float bv = bias[gn];
#pragma unroll
      for (int r = 0; r < 8; ++r) {
        const int gm = m0 + wm + mt * 16 + hi * 8 + r;
        float v = acc[mt][nt][r] + bv;
        if (EPI == EPI_BIAS_GELU_BF16)
          v = 0.5f * v * (1.f + erff(v * 0.70710678118654752f));
        if (EPI == EPI_BIAS_RES_F32) {
          v += resid[(size_t)gm * N + gn];
          reinterpret_cast<float*>(out)[(size_t)gm * N + gn] = v;
        } else {
          reinterpret_cast<bf16*>(out)[(size_t)gm * N + gn] = (bf16)v;
        }
      }
    }
}

// ---------------------------------------------------------------- attention
// Flash-attention, causal. grid = (T/128, B*H). 8 waves/block, 16 q-rows/wave.
__global__ __launch_bounds__(256) void attn_kernel(
    const bf16* __restrict__ qkv,   // [B*T][2304]  (q|k|v)
    bf16* __restrict__ y)           // [B*T][768]
{
  const int T = 2048, HD = 64, QS = 2304, NH = 12;
  const int bh = blockIdx.y;
  const int b  = bh / NH, h = bh % NH;
  const int tid = threadIdx.x, wid = tid >> 5, lane = tid & 31;
  const int l = lane & 15, hi = lane >> 4;
  const int q0 = blockIdx.x * 128;
  const int qw = q0 + wid * 16;

  const bf16* Qp = qkv + (size_t)b * T * QS + h * HD;
  const bf16* Kp = Qp + 768;
  const bf16* Vp = Qp + 1536;

  __shared__ __align__(16) bf16 Ks[64][64];     // [key][d]   (async staged)
  __shared__ __align__(16) bf16 Vt[64][64];     // [d][key]
  __shared__ __align__(16) bf16 Pl[8][16][64];  // wave-private P relayout slab

  // Q resident in registers as two A-fragments (d = 0..31, 32..63)
  FragBF qf[2];
  {
    const bf16* qr = Qp + (size_t)(qw + l) * QS;
#pragma unroll
    for (int f = 0; f < 2; ++f) {
      qf[f].u[0] = *reinterpret_cast<const uint4*>(qr + f * 32 + hi * 8);
      qf[f].u[1] = *reinterpret_cast<const uint4*>(qr + f * 32 + 16 + hi * 8);
    }
  }

  v8f o[4] = {};
  float mrow[8], lrow[8];
#pragma unroll
  for (int r = 0; r < 8; ++r) { mrow[r] = -1e30f; lrow[r] = 0.f; }

  const int ktmax = (q0 + 127) >> 6;
  const int kvr = tid >> 2, kvc = (tid & 3) * 16;
  const uint32_t ldsK = lds_addr_of(&Ks[kvr][kvc]);

  for (int kt = 0; kt <= ktmax; ++kt) {
    const int k0 = kt * 64;
    {  // K tile: async DMA straight into LDS
      const bf16* ks = Kp + (size_t)(k0 + kvr) * QS + kvc;
      async_ld_b128(ldsK,      ks);
      async_ld_b128(ldsK + 16, ks + 8);
      // V tile: transposed scatter (2-byte granularity, stays on vector path)
      const uint4* sv = reinterpret_cast<const uint4*>(Vp + (size_t)(k0 + kvr) * QS + kvc);
      FragBF t; t.u[0] = sv[0]; t.u[1] = sv[1];
#pragma unroll
      for (int i = 0; i < 16; ++i) Vt[kvc + i][kvr] = t.e[i];
    }
    wait_async0();
    __syncthreads();

    if (k0 <= qw + 15) {          // wave-uniform: EXEC all-ones for WMMA
      // S = Q @ K^T
      v8f s[4] = {};
#pragma unroll
      for (int nt = 0; nt < 4; ++nt) {
        FragBF kb[2];
        const bf16* p = &Ks[nt * 16 + l][0];
#pragma unroll
        for (int f = 0; f < 2; ++f) {
          kb[f].u[0] = reinterpret_cast<const uint4*>(p + f * 32 + hi * 8)[0];
          kb[f].u[1] = reinterpret_cast<const uint4*>(p + f * 32 + hi * 8)[2];
        }
        s[nt] = wmma_bf16(qf[0].v, kb[0].v, s[nt]);
        s[nt] = wmma_bf16(qf[1].v, kb[1].v, s[nt]);
      }
      // scale + causal mask + online softmax (16-lane-half row reductions)
#pragma unroll
      for (int r = 0; r < 8; ++r) {
        const int q = qw + hi * 8 + r;
        float rmax = -1e30f;
#pragma unroll
        for (int nt = 0; nt < 4; ++nt) {
          const int key = k0 + nt * 16 + l;
          float v = s[nt][r] * 0.125f;          // 1/sqrt(64)
          if (key > q) v = -1e30f;
          s[nt][r] = v;
          rmax = fmaxf(rmax, v);
        }
#pragma unroll
        for (int off = 1; off < 16; off <<= 1)
          rmax = fmaxf(rmax, __shfl_xor(rmax, off, 32));
        const float mnew  = fmaxf(mrow[r], rmax);
        const float alpha = __expf(mrow[r] - mnew);
        float rsum = 0.f;
#pragma unroll
        for (int nt = 0; nt < 4; ++nt) {
          const float p = __expf(s[nt][r] - mnew);
          s[nt][r] = p;
          rsum += p;
        }
#pragma unroll
        for (int off = 1; off < 16; off <<= 1)
          rsum += __shfl_xor(rsum, off, 32);
        lrow[r] = lrow[r] * alpha + rsum;
        mrow[r] = mnew;
#pragma unroll
        for (int nt = 0; nt < 4; ++nt) o[nt][r] *= alpha;
      }
      // re-layout P: C-layout -> LDS -> A-fragments (wave-private, in-order LDS)
#pragma unroll
      for (int nt = 0; nt < 4; ++nt)
#pragma unroll
        for (int r = 0; r < 8; ++r)
          Pl[wid][hi * 8 + r][nt * 16 + l] = (bf16)s[nt][r];
      FragBF pf[2];
#pragma unroll
      for (int f = 0; f < 2; ++f) {
        const bf16* p = &Pl[wid][l][f * 32];
        pf[f].u[0] = *reinterpret_cast<const uint4*>(p + hi * 8);
        pf[f].u[1] = *reinterpret_cast<const uint4*>(p + 16 + hi * 8);
      }
      // O += P @ V
#pragma unroll
      for (int nt = 0; nt < 4; ++nt) {
        FragBF vb[2];
        const bf16* p = &Vt[nt * 16 + l][0];
#pragma unroll
        for (int f = 0; f < 2; ++f) {
          vb[f].u[0] = reinterpret_cast<const uint4*>(p + f * 32 + hi * 8)[0];
          vb[f].u[1] = reinterpret_cast<const uint4*>(p + f * 32 + hi * 8)[2];
        }
        o[nt] = wmma_bf16(pf[0].v, vb[0].v, o[nt]);
        o[nt] = wmma_bf16(pf[1].v, vb[1].v, o[nt]);
      }
    }
    __syncthreads();
  }

  // normalize + write y (bf16 feeds proj GEMM)
#pragma unroll
  for (int nt = 0; nt < 4; ++nt)
#pragma unroll
    for (int r = 0; r < 8; ++r) {
      const int q = qw + hi * 8 + r;
      const float v = o[nt][r] / lrow[r];
      y[(size_t)(b * T + q) * 768 + h * HD + nt * 16 + l] = (bf16)v;
    }
}

// ---------------------------------------------------------------- host
extern "C" void kernel_launch(void* const* d_in, const int* in_sizes, int n_in,
                              void* d_out, int out_size, void* d_ws, size_t ws_size,
                              hipStream_t stream) {
  const float* x     = (const float*)d_in[0];
  const float* ln1w  = (const float*)d_in[1];
  const float* ln1b  = (const float*)d_in[2];
  const float* wqkv  = (const float*)d_in[3];
  const float* bqkv  = (const float*)d_in[4];
  const float* wproj = (const float*)d_in[5];
  const float* bproj = (const float*)d_in[6];
  const float* ln2w  = (const float*)d_in[7];
  const float* ln2b  = (const float*)d_in[8];
  const float* wfc1  = (const float*)d_in[9];
  const float* bfc1  = (const float*)d_in[10];
  const float* wfc2  = (const float*)d_in[11];
  const float* bfc2  = (const float*)d_in[12];

  const int M = 4096, C = 768;          // M = B*T
  char* ws = (char*)d_ws;
  size_t off = 0;
  auto alloc = [&](size_t bytes) -> void* {
    void* p = ws + off;
    off = (off + bytes + 255) & ~(size_t)255;
    return p;
  };
  // transposed bf16 weights: [N][K]
  bf16* wqkv_t = (bf16*)alloc((size_t)3 * C * C * 2);
  bf16* wprj_t = (bf16*)alloc((size_t)C * C * 2);
  bf16* wfc1_t = (bf16*)alloc((size_t)4 * C * C * 2);
  bf16* wfc2_t = (bf16*)alloc((size_t)C * 4 * C * 2);
  bf16* h1     = (bf16*)alloc((size_t)M * C * 2);
  bf16* qkv_b  = (bf16*)alloc((size_t)M * 3 * C * 2);
  bf16* y_b    = (bf16*)alloc((size_t)M * C * 2);
  float* x1    = (float*)alloc((size_t)M * C * 4);
  bf16* h2     = (bf16*)alloc((size_t)M * C * 2);
  bf16* act_b  = (bf16*)alloc((size_t)M * 4 * C * 2);

  // one-time weight convert+transpose (f32[K][N] -> bf16[N][K])
  transpose_bf16_kernel<<<dim3(3 * C / 32, C / 32), 256, 0, stream>>>(wqkv, wqkv_t, C, 3 * C);
  transpose_bf16_kernel<<<dim3(C / 32, C / 32), 256, 0, stream>>>(wproj, wprj_t, C, C);
  transpose_bf16_kernel<<<dim3(4 * C / 32, C / 32), 256, 0, stream>>>(wfc1, wfc1_t, C, 4 * C);
  transpose_bf16_kernel<<<dim3(C / 32, 4 * C / 32), 256, 0, stream>>>(wfc2, wfc2_t, 4 * C, C);

  // h1 = LN1(x)
  layernorm_kernel<<<M, 256, 0, stream>>>(x, ln1w, ln1b, h1, C);
  // qkv = h1 @ w_qkv + b_qkv
  gemm_bf16_kernel<EPI_BIAS_BF16><<<dim3(3 * C / 128, M / 128), 256, 0, stream>>>(
      h1, wqkv_t, bqkv, nullptr, qkv_b, M, 3 * C, C);
  // y = causal softmax attention
  attn_kernel<<<dim3(2048 / 128, 2 * 12), 256, 0, stream>>>(qkv_b, y_b);
  // x1 = x + y @ w_proj + b_proj
  gemm_bf16_kernel<EPI_BIAS_RES_F32><<<dim3(C / 128, M / 128), 256, 0, stream>>>(
      y_b, wprj_t, bproj, x, x1, M, C, C);
  // h2 = LN2(x1)
  layernorm_kernel<<<M, 256, 0, stream>>>(x1, ln2w, ln2b, h2, C);
  // act = gelu(h2 @ w_fc1 + b_fc1)
  gemm_bf16_kernel<EPI_BIAS_GELU_BF16><<<dim3(4 * C / 128, M / 128), 256, 0, stream>>>(
      h2, wfc1_t, bfc1, nullptr, act_b, M, 4 * C, C);
  // out = x1 + act @ w_fc2 + b_fc2
  gemm_bf16_kernel<EPI_BIAS_RES_F32><<<dim3(C / 128, M / 128), 256, 0, stream>>>(
      act_b, wfc2_t, bfc2, x1, (float*)d_out, M, C, 4 * C);
}
